// E3ProbeMessageModel_3315714752867
// MI455X (gfx1250) — compile-verified, matrix operand order checked by reading, not run on hardware
//
#include <hip/hip_runtime.h>
#include <hip/hip_bf16.h>

typedef __attribute__((ext_vector_type(16))) __bf16   v16bf;
typedef __attribute__((ext_vector_type(8)))  float    v8f;
typedef __attribute__((ext_vector_type(8)))  unsigned v8u;

#define A_NODES 2000
#define P_NODES 20000
#define E_EDGES 200000
#define MUL0 128
#define MUL1 64
#define DIM 320
#define NBASIS 10
#define FC_HID 100
#define WNUM 384
#define MID0 192
#define RWID 768   // accumulator row: out0a(128)|out0b(64)|out1a(384)|out1b(192)
#define OWID 320

#define INV_SQRT10  0.31622776601683794f
#define INV_SQRT100 0.1f
#define INV_SQRT128 0.08838834764831845f
#define INV_SQRT64  0.125f
#define INV_SQRT192 0.07216878364870323f
#define INV_SQRT3   0.5773502691896258f

static __device__ __forceinline__ unsigned bfbits(float f) {
  unsigned u = __float_as_uint(f);
  return (u + 0x7FFFu + ((u >> 16) & 1u)) >> 16;     // RNE bf16 bits
}
static __device__ __forceinline__ __bf16 f2bf(float f) {
  unsigned short h = (unsigned short)bfbits(f);
  return __builtin_bit_cast(__bf16, h);
}

// ---- CDNA5 wave32 WMMA fragment-major layout helpers ----
// lanes 0-15 hold k_local {0..7,16..23}, lanes 16-31 hold {8..15,24..31};
// 16 bf16 elements per lane per 32-K tile.
static __device__ __forceinline__ int frag_half(int kl) { return (kl >> 3) & 1; }
static __device__ __forceinline__ int frag_elem(int kl) { return (kl & 7) + ((kl >> 4) & 1) * 8; }

static __device__ __forceinline__ void scatter_bfrag(__bf16* dst, int N, int k, int n, float val) {
  int ntiles = N >> 4;
  int kl = k & 31;
  int lane = (n & 15) + 16 * frag_half(kl);
  dst[((((k >> 5) * ntiles + (n >> 4)) * 32 + lane) << 4) + frag_elem(kl)] = f2bf(val);
}
static __device__ __forceinline__ void scatter_afrag(__bf16* dst, int m, int k, float val) {
  int kl = k & 31;
  int lane = m + 16 * frag_half(kl);
  dst[((((k >> 5) * 32) + lane) << 4) + frag_elem(kl)] = f2bf(val);
}

static __device__ __forceinline__ v8f wmma_bf16(v16bf a, v16bf b, v8f c) {
  return __builtin_amdgcn_wmma_f32_16x16x32_bf16(false, a, false, b, (short)0, c, false, false);
}
static __device__ __forceinline__ float fast_silu(float x) {
  return x * __builtin_amdgcn_rcpf(1.f + __expf(-x));
}

// ------------------------------------------------------------------
// Pre-swizzle all reused weight matrices into fragment-major bf16, with the
// reference's normalization constants folded in (runs once, tiny).
__global__ void prep_weights(const float* fc_w1, const float* fc_w2,
                             const float* l2w0, const float* l2w1,
                             const float* scw0, const float* scw1,
                             __bf16* fcw1v, __bf16* fcw2v,
                             __bf16* l2w0v, __bf16* l2w1v,
                             __bf16* scw0v, __bf16* scw1v) {
  int tid = blockIdx.x * blockDim.x + threadIdx.x;
  int nth = gridDim.x * blockDim.x;
  for (int i = tid; i < 32 * 112; i += nth) {          // fc_w1 [10,100] -> padded [32,112]
    int k = i / 112, n = i % 112;
    float v = (k < NBASIS && n < FC_HID) ? fc_w1[k * FC_HID + n] * INV_SQRT10 : 0.f;
    scatter_bfrag(fcw1v, 112, k, n, v);
  }
  for (int i = tid; i < 128 * WNUM; i += nth) {        // fc_w2 [100,384] -> padded [128,384]
    int k = i / WNUM, n = i % WNUM;
    scatter_bfrag(fcw2v, WNUM, k, n, (k < FC_HID) ? fc_w2[k * WNUM + n] * INV_SQRT100 : 0.f);
  }
  for (int i = tid; i < MID0 * MUL0; i += nth)
    scatter_bfrag(l2w0v, MUL0, i / MUL0, i % MUL0, l2w0[i] * INV_SQRT192);
  for (int i = tid; i < MID0 * MUL1; i += nth)
    scatter_bfrag(l2w1v, MUL1, i / MUL1, i % MUL1, l2w1[i] * INV_SQRT192);
  for (int i = tid; i < MUL0 * MUL0; i += nth)
    scatter_bfrag(scw0v, MUL0, i / MUL0, i % MUL0, scw0[i] * INV_SQRT128);
  for (int i = tid; i < MUL1 * MUL1; i += nth)
    scatter_bfrag(scw1v, MUL1, i / MUL1, i % MUL1, scw1[i] * INV_SQRT64);
}

// ------------------------------------------------------------------
// Sender-side linears (A=2000 rows, 114 MFLOP total: VALU, stays in L2)
__global__ void sender_linear(const float* s_in, const float* s_attr,
                              const float* w0, const float* w1,
                              float* s0, float* s1) {
  int tid = blockIdx.x * blockDim.x + threadIdx.x;
  if (tid >= A_NODES * 320) return;
  int a = tid / 320, j = tid % 320;
  float attr = s_attr[a];
  const float* row = s_in + (size_t)a * DIM;
  if (j < MUL0) {
    float acc = 0.f;
    for (int u = 0; u < MUL0; ++u) acc += row[u] * w0[u * MUL0 + j];
    s0[(size_t)a * MUL0 + j] = acc * attr * INV_SQRT128;
  } else {
    int j2 = j - MUL0, v = j2 / 3, m = j2 % 3;
    float acc = 0.f;
    for (int u = 0; u < MUL1; ++u) acc += row[MUL0 + u * 3 + m] * w1[u * MUL1 + v];
    s1[(size_t)a * 192 + j2] = acc * attr * INV_SQRT64;   // layout [a][v*3+m]
  }
}

// ------------------------------------------------------------------
// Fused edge kernel: 1 wave = 16 edges. FC MLP via WMMA, tensor-product
// messages scattered with f32 atomics (accumulator lives in the 192MB L2).
__global__ __launch_bounds__(32) void edge_kernel(
    const float* __restrict__ edge_scalars, const int* __restrict__ edge_src,
    const int* __restrict__ edge_dst, const float* __restrict__ edge_attr,
    const v16bf* __restrict__ fcw1v, const v16bf* __restrict__ fcw2v,
    const float* __restrict__ s0, const float* __restrict__ s1,
    float* __restrict__ r_acc) {
  __shared__ v16bf h_frag[4 * 32];     // hidden acts in A-fragment layout (K padded to 128)
  __shared__ float w_lds[16 * WNUM];   // per-edge weights
  int lane  = threadIdx.x;
  int e0    = blockIdx.x * 16;
  int mrow  = lane & 15;
  int kb    = (lane < 16) ? 0 : 8;
  int nlane = lane & 15;
  int mbase = (lane >> 4) * 8;

  // prefetch next tile's edge data while we compute this one
  if (e0 + 16 < E_EDGES) {
    __builtin_prefetch(edge_scalars + (size_t)(e0 + 16 + mrow) * NBASIS, 0, 1);
    __builtin_prefetch(edge_attr + (size_t)(e0 + 16) * 4 + lane, 0, 1);
  }

  // A fragment of edge_scalars (16x10 padded to 16x32), packed as u32 pairs
  v16bf a;
  {
    const float* row = edge_scalars + (size_t)(e0 + mrow) * NBASIS;
    v8u ai;
#pragma unroll
    for (int j = 0; j < 8; ++j) {
      int k = (j < 4) ? (kb + 2 * j) : (kb + 8 + 2 * j);
      unsigned lo = (k     < NBASIS) ? bfbits(row[k])     : 0u;
      unsigned hi = (k + 1 < NBASIS) ? bfbits(row[k + 1]) : 0u;
      ai[j] = lo | (hi << 16);
    }
    a = __builtin_bit_cast(v16bf, ai);
  }
  // zero h fragment store (covers the K=112..127 pad)
  {
    unsigned long long* hz = (unsigned long long*)h_frag;
#pragma unroll
    for (int i = 0; i < 16; ++i) hz[lane + 32 * i] = 0ull;
  }

  // GEMM1 (K=32 single step, 7 N-tiles) + silu -> h_frag (A-fragment layout).
  // 1/sqrt(NBASIS) already folded into fcw1v.
#pragma unroll
  for (int t = 0; t < 7; ++t) {
    v16bf b = fcw1v[t * 32 + lane];
    v8f c = {};
    c = wmma_bf16(a, b, c);
#pragma unroll
    for (int rr = 0; rr < 8; ++rr)
      scatter_afrag((__bf16*)h_frag, mbase + rr, t * 16 + nlane, fast_silu(c[rr]));
  }
  __syncthreads();

  // GEMM2: weight = h @ fc_w2 (scale folded), K=128 (4 steps), N=384 (24 tiles)
  v16bf ha[4];
#pragma unroll
  for (int kk = 0; kk < 4; ++kk) ha[kk] = h_frag[kk * 32 + lane];
#pragma unroll 1
  for (int t = 0; t < 24; ++t) {
    v8f acc = {};
#pragma unroll
    for (int kk = 0; kk < 4; ++kk) {
      v16bf b = fcw2v[(kk * 24 + t) * 32 + lane];
      acc = wmma_bf16(ha[kk], b, acc);
    }
#pragma unroll
    for (int rr = 0; rr < 8; ++rr)
      w_lds[(mbase + rr) * WNUM + t * 16 + nlane] = acc[rr];
  }
  __syncthreads();

  // Tensor-product messages + scatter-add (lane-parallel, contiguous channels)
  for (int i = 0; i < 16; ++i) {
    int e = e0 + i;
    int src = edge_src[e];
    int dst = edge_dst[e];
    float sh0  = edge_attr[(size_t)e * 4 + 0];
    float sh1x = edge_attr[(size_t)e * 4 + 1];
    float sh1y = edge_attr[(size_t)e * 4 + 2];
    float sh1z = edge_attr[(size_t)e * 4 + 3];
    const float* g0 = s0 + (size_t)src * MUL0;
    const float* g1 = s1 + (size_t)src * 192;
    float* rp = r_acc + (size_t)dst * RWID;
    const float* wr = w_lds + i * WNUM;
#pragma unroll
    for (int t = 0; t < 4; ++t) {                 // out0a: w0*g0*sh0
      int c = lane + 32 * t;
      unsafeAtomicAdd(rp + c, wr[c] * g0[c] * sh0);
    }
#pragma unroll
    for (int t = 0; t < 2; ++t) {                 // out0b: w3*(g1.sh1)/sqrt3
      int c = lane + 32 * t;
      float dot = g1[c * 3] * sh1x + g1[c * 3 + 1] * sh1y + g1[c * 3 + 2] * sh1z;
      unsafeAtomicAdd(rp + 128 + c, wr[320 + c] * dot * INV_SQRT3);
    }
#pragma unroll
    for (int t = 0; t < 12; ++t) {                // out1a: w1*g0 x sh1
      int idx = lane + 32 * t;
      int u = idx / 3, mm = idx - 3 * u;
      float sh = (mm == 0) ? sh1x : ((mm == 1) ? sh1y : sh1z);
      unsafeAtomicAdd(rp + 192 + idx, wr[128 + u] * g0[u] * sh);
    }
#pragma unroll
    for (int t = 0; t < 6; ++t) {                 // out1b: w2*g1*sh0
      int idx = lane + 32 * t;
      int v = idx / 3, mm = idx - 3 * v;
      unsafeAtomicAdd(rp + 576 + idx, wr[256 + v] * g1[v * 3 + mm] * sh0);
    }
  }
}

// ------------------------------------------------------------------
// Receiver kernel: 1 wave = 16 receivers; conv/self-connection GEMMs via
// WMMA with fragment-major LDS staging, then the cos/sin gate.
__global__ __launch_bounds__(32) void receiver_kernel(
    const float* __restrict__ r_acc, const float* __restrict__ recv_in,
    const float* __restrict__ recv_attr,
    const v16bf* __restrict__ l2w0v, const v16bf* __restrict__ l2w1v,
    const v16bf* __restrict__ scw0v, const v16bf* __restrict__ scw1v,
    const float* __restrict__ lin3_w, float* __restrict__ out) {
  __shared__ v16bf r0f[6 * 32];        // r0 tile, A-frag layout, K=192
  __shared__ v16bf r1f[3][6 * 32];     // r1 per component
  __shared__ v16bf rin0f[4 * 32];      // receiver_input scalar part, K=128
  __shared__ v16bf rin1f[3][2 * 32];   // receiver_input vector part, K=64
  __shared__ float cosA[16], sinA[16];
  int lane = threadIdx.x;
  int p0   = blockIdx.x * 16;

  for (int i = lane; i < 16 * 192; i += 32) {
    int row = i / 192, c = i % 192;
    int p = p0 + row;
    float ra = recv_attr[p] * INV_SQRT10;          // 1/sqrt(NUM_NEIGHBORS)
    scatter_afrag((__bf16*)r0f, row, c, r_acc[(size_t)p * RWID + c] * ra);
#pragma unroll
    for (int mm = 0; mm < 3; ++mm)
      scatter_afrag((__bf16*)r1f[mm], row, c,
                    r_acc[(size_t)p * RWID + 192 + c * 3 + mm] * ra);
  }
  for (int i = lane; i < 16 * 128; i += 32) {
    int row = i / 128, c = i % 128;
    int p = p0 + row;
    scatter_afrag((__bf16*)rin0f, row, c, recv_in[(size_t)p * DIM + c] * recv_attr[p]);
  }
  for (int i = lane; i < 16 * 64; i += 32) {
    int row = i / 64, c = i % 64;
    int p = p0 + row;
    float ra = recv_attr[p];
#pragma unroll
    for (int mm = 0; mm < 3; ++mm)
      scatter_afrag((__bf16*)rin1f[mm], row, c,
                    recv_in[(size_t)p * DIM + 128 + c * 3 + mm] * ra);
  }

  if (lane < 16) {                                 // angle gate per receiver (f32)
    int p = p0 + lane;
    float acc = 0.f;
    for (int u = 0; u < 192; ++u) acc += r_acc[(size_t)p * RWID + u] * lin3_w[u];
    float ang = 0.1f * acc * recv_attr[p] * INV_SQRT10 * INV_SQRT192;
    cosA[lane] = __cosf(ang);
    sinA[lane] = __sinf(ang);
  }
  __syncthreads();

  int nlane = lane & 15, mbase = (lane >> 4) * 8;
  // scalar channels: conv0 (K=192) + sc0 (K=128), 8 N-tiles of 16.
  // All normalization folded into the bf16 weights -> mul+fma epilogue.
#pragma unroll 1
  for (int t = 0; t < 8; ++t) {
    v8f conv = {};
#pragma unroll
    for (int kk = 0; kk < 6; ++kk)
      conv = wmma_bf16(r0f[kk * 32 + lane], l2w0v[(kk * 8 + t) * 32 + lane], conv);
    v8f sc = {};
#pragma unroll
    for (int kk = 0; kk < 4; ++kk)
      sc = wmma_bf16(rin0f[kk * 32 + lane], scw0v[(kk * 8 + t) * 32 + lane], sc);
#pragma unroll
    for (int rr = 0; rr < 8; ++rr) {
      int row = mbase + rr;
      out[(size_t)(p0 + row) * OWID + t * 16 + nlane] =
          __fmaf_rn(cosA[row], sc[rr], sinA[row] * conv[rr]);
    }
  }
  // vector channels: per m, conv1 (K=192) + sc1 (K=64), 4 N-tiles of 16
#pragma unroll 1
  for (int mm = 0; mm < 3; ++mm) {
#pragma unroll 1
    for (int t = 0; t < 4; ++t) {
      v8f conv = {};
#pragma unroll
      for (int kk = 0; kk < 6; ++kk)
        conv = wmma_bf16(r1f[mm][kk * 32 + lane], l2w1v[(kk * 4 + t) * 32 + lane], conv);
      v8f sc = {};
#pragma unroll
      for (int kk = 0; kk < 2; ++kk)
        sc = wmma_bf16(rin1f[mm][kk * 32 + lane], scw1v[(kk * 4 + t) * 32 + lane], sc);
#pragma unroll
      for (int rr = 0; rr < 8; ++rr) {
        int row = mbase + rr;
        out[(size_t)(p0 + row) * OWID + 128 + (t * 16 + nlane) * 3 + mm] =
            __fmaf_rn(cosA[row], sc[rr], sinA[row] * conv[rr]);
      }
    }
  }
}

// ------------------------------------------------------------------
extern "C" void kernel_launch(void* const* d_in, const int* in_sizes, int n_in,
                              void* d_out, int out_size, void* d_ws, size_t ws_size,
                              hipStream_t stream) {
  const float* sender_input   = (const float*)d_in[0];
  const float* sender_attr    = (const float*)d_in[1];
  const float* receiver_input = (const float*)d_in[2];
  const float* receiver_attr  = (const float*)d_in[3];
  const int*   edge_src       = (const int*)d_in[4];
  const int*   edge_dst       = (const int*)d_in[5];
  const float* edge_attr      = (const float*)d_in[6];
  const float* edge_scalars   = (const float*)d_in[7];
  const float* fc_w1   = (const float*)d_in[8];
  const float* fc_w2   = (const float*)d_in[9];
  const float* lin1_w0 = (const float*)d_in[10];
  const float* lin1_w1 = (const float*)d_in[11];
  const float* sc_w0   = (const float*)d_in[12];
  const float* sc_w1   = (const float*)d_in[13];
  const float* lin2_w0 = (const float*)d_in[14];
  const float* lin2_w1 = (const float*)d_in[15];
  const float* lin3_w  = (const float*)d_in[16];

  char* ws = (char*)d_ws;
  size_t off = 0;
  auto alloc = [&](size_t bytes) -> void* {
    void* p = ws + off;
    off += (bytes + 255) & ~(size_t)255;
    return p;
  };
  float*  r_acc = (float*)alloc((size_t)P_NODES * RWID * sizeof(float));
  float*  s0    = (float*)alloc((size_t)A_NODES * MUL0 * sizeof(float));
  float*  s1    = (float*)alloc((size_t)A_NODES * 192 * sizeof(float));
  __bf16* fcw1v = (__bf16*)alloc(32 * 112 * sizeof(__bf16));
  __bf16* fcw2v = (__bf16*)alloc(128 * WNUM * sizeof(__bf16));
  __bf16* l2w0v = (__bf16*)alloc(MID0 * MUL0 * sizeof(__bf16));
  __bf16* l2w1v = (__bf16*)alloc(MID0 * MUL1 * sizeof(__bf16));
  __bf16* scw0v = (__bf16*)alloc(MUL0 * MUL0 * sizeof(__bf16));
  __bf16* scw1v = (__bf16*)alloc(MUL1 * MUL1 * sizeof(__bf16));

  hipMemsetAsync(r_acc, 0, (size_t)P_NODES * RWID * sizeof(float), stream);
  prep_weights<<<128, 256, 0, stream>>>(fc_w1, fc_w2, lin2_w0, lin2_w1, sc_w0, sc_w1,
                                        fcw1v, fcw2v, l2w0v, l2w1v, scw0v, scw1v);
  sender_linear<<<(A_NODES * 320 + 255) / 256, 256, 0, stream>>>(
      sender_input, sender_attr, lin1_w0, lin1_w1, s0, s1);
  edge_kernel<<<E_EDGES / 16, 32, 0, stream>>>(
      edge_scalars, edge_src, edge_dst, edge_attr,
      (const v16bf*)fcw1v, (const v16bf*)fcw2v, s0, s1, r_acc);
  receiver_kernel<<<P_NODES / 16, 32, 0, stream>>>(
      r_acc, receiver_input, receiver_attr,
      (const v16bf*)l2w0v, (const v16bf*)l2w1v,
      (const v16bf*)scw0v, (const v16bf*)scw1v, lin3_w, (float*)d_out);
}
